// BetaE_70841190580392
// MI455X (gfx1250) — matrix operand order repaired and checked by reading, not compiled, not run on hardware
//
#include <hip/hip_runtime.h>
#include <hip/hip_bf16.h>

typedef __attribute__((ext_vector_type(16))) __bf16 bf16x16;
typedef __attribute__((ext_vector_type(8)))  __bf16 bf16x8;
typedef __attribute__((ext_vector_type(8)))  float  f32x8;

#define BM 128
#define BN 128
#define BK 32

// async global -> LDS 16-byte copy (CDNA5 GLOBAL_LOAD_ASYNC_TO_LDS_B128, GVS mode)
#define ASYNC_LOAD_B128(ldsoff, voff, sbase)                                \
    asm volatile("global_load_async_to_lds_b128 %0, %1, %2"                 \
                 :: "v"(ldsoff), "v"(voff), "s"(sbase) : "memory")

__device__ __forceinline__ void wait_asynccnt_4() {
#if __has_builtin(__builtin_amdgcn_s_wait_asynccnt)
    __builtin_amdgcn_s_wait_asynccnt(4);
#else
    asm volatile("s_wait_asynccnt 0x4" ::: "memory");
#endif
}
__device__ __forceinline__ void wait_asynccnt_0() {
#if __has_builtin(__builtin_amdgcn_s_wait_asynccnt)
    __builtin_amdgcn_s_wait_asynccnt(0);
#else
    asm volatile("s_wait_asynccnt 0x0" ::: "memory");
#endif
}

// ---------------------------------------------------------------------------
// Gather + entity regularizer + concat -> bf16 [16384 x 1216] (K padded)
// ---------------------------------------------------------------------------
__global__ __launch_bounds__(256) void prep_kernel(
    const int* __restrict__ a1, const int* __restrict__ r1,
    const int* __restrict__ a2, const int* __restrict__ r2,
    const float* __restrict__ ent, const float* __restrict__ rel,
    __bf16* __restrict__ Xc)
{
    const int row = blockIdx.x;
    int eidx, ridx;
    if (row < 8192) { eidx = a1[row];        ridx = r1[row]; }
    else            { eidx = a2[row - 8192]; ridx = r2[row - 8192]; }
    const float* e = ent + (size_t)eidx * 800;
    const float* r = rel + (size_t)ridx * 400;
    __bf16* dst = Xc + (size_t)row * 1216;
    for (int c = threadIdx.x; c < 1216; c += 256) {
        float v;
        if (c < 800) {
            v = e[c] + 1.0f;                       // entity regularizer
            v = fminf(fmaxf(v, 0.05f), 1e9f);
        } else if (c < 1200) {
            v = r[c - 800];
        } else {
            v = 0.0f;                              // K padding
        }
        dst[c] = (__bf16)v;
    }
}

// ---------------------------------------------------------------------------
// Weight f32 [N,K] -> bf16 [N,Kp] with zero K-padding
// ---------------------------------------------------------------------------
__global__ __launch_bounds__(256) void wconv_kernel(
    const float* __restrict__ src, __bf16* __restrict__ dst, int K, int Kp)
{
    const int row = blockIdx.x;
    for (int c = threadIdx.x; c < Kp; c += 256) {
        float v = (c < K) ? src[(size_t)row * K + c] : 0.0f;
        dst[(size_t)row * Kp + c] = (__bf16)v;
    }
}

// ---------------------------------------------------------------------------
// GEMM: out = epilogue(A[M,K](bf16) * B[N,K]^T(bf16) + bias)
// 128x128 block, 8 waves (2Mx4N), wave tile 64x32 = 4x2 WMMA tiles, BK=32.
// Tiles staged via async global->LDS DMA, double-buffered, 2 tiles in flight.
// mode 0: relu -> bf16   mode 1: reg(clip(x+1,.05,1e9)) -> f32+bf16
// mode 2: plain -> f32
// ---------------------------------------------------------------------------
__global__ __launch_bounds__(256) void gemm_bf16_wmma(
    const __bf16* __restrict__ A, int lda,
    const __bf16* __restrict__ B, int ldb,
    const float* __restrict__ bias,
    int N, int K, int mode,
    __bf16* __restrict__ outB, float* __restrict__ outF, int ldo)
{
    __shared__ __attribute__((aligned(16))) __bf16 As[2][BM * BK];
    __shared__ __attribute__((aligned(16))) __bf16 Bs[2][BN * BK];

    const int tid   = threadIdx.x;
    const int lane  = tid & 31;
    const int wv    = tid >> 5;       // 0..7
    const int waveM = wv >> 2;        // 0..1  (rows of 64)
    const int waveN = wv & 3;         // 0..3  (cols of 32)
    const int m0 = blockIdx.y * BM;
    const int n0 = blockIdx.x * BN;
    const int am = lane & 15;         // M / N index within 16-wide tile
    const int ks = lane >> 4;         // K half-select

    // issue one tile pair (A 128x32 + B 128x32) as 4 async b128 ops/thread.
    // B rows are clamped (not zeroed): OOB rows only feed output columns >= N
    // which the epilogue never stores.
    auto issue_tile = [&](int buf, int k0) {
        const __bf16* abase = A + (size_t)m0 * lda + k0;   // uniform -> SGPR
        const __bf16* bbase = B + k0;                      // uniform -> SGPR
#pragma unroll
        for (int i = 0; i < 2; ++i) {
            int cidx = tid + i * 256;            // 0..511 chunks of 8 bf16
            int r  = cidx >> 2;
            int ce = (cidx & 3) * 8;
            unsigned av = (unsigned)((r * lda + ce) * (int)sizeof(__bf16));
            unsigned al = (unsigned)(uintptr_t)&As[buf][r * BK + ce];
            ASYNC_LOAD_B128(al, av, abase);
            int nrow = n0 + r;
            nrow = (nrow < N - 1) ? nrow : (N - 1);
            unsigned bv = (unsigned)(((size_t)nrow * ldb + ce) * sizeof(__bf16));
            unsigned bl = (unsigned)(uintptr_t)&Bs[buf][r * BK + ce];
            ASYNC_LOAD_B128(bl, bv, bbase);
        }
    };

    f32x8 acc[4][2];
    const f32x8 zero = {0.f, 0.f, 0.f, 0.f, 0.f, 0.f, 0.f, 0.f};
#pragma unroll
    for (int i = 0; i < 4; ++i)
#pragma unroll
        for (int j = 0; j < 2; ++j) acc[i][j] = zero;

    issue_tile(0, 0);                 // prologue: tile 0 in flight
    int cur = 0;
    for (int k0 = 0; k0 < K; k0 += BK) {
        const bool more = (k0 + BK) < K;
        if (more) issue_tile(cur ^ 1, k0 + BK);   // next tile in flight
        if (more) wait_asynccnt_4();              // current tile landed
        else      wait_asynccnt_0();
        __syncthreads();

        // ---- B fragments (32x16 K-major, contiguous k = 16*ks + s) ------------
        union Frag { bf16x16 v; bf16x8 h[2]; };
        Frag bfrag[2];
#pragma unroll
        for (int nt = 0; nt < 2; ++nt) {
            int brow = waveN * 32 + nt * 16 + am;
            const bf16x8* bp = (const bf16x8*)&Bs[cur][brow * BK];
            bfrag[nt].h[0] = bp[ks * 2];          // K = ks*16 .. +8
            bfrag[nt].h[1] = bp[ks * 2 + 1];      // K = ks*16+8 .. +8
        }
        // ---- A fragments + WMMA ------------------------------------------
#pragma unroll
        for (int mt = 0; mt < 4; ++mt) {
            int arow = waveM * 64 + mt * 16 + am;
            const bf16x8* ap = (const bf16x8*)&As[cur][arow * BK];
            Frag af;
            af.h[0] = ap[ks * 2];                 // K = ks*16 .. +8
            af.h[1] = ap[ks * 2 + 1];             // K = ks*16+8 .. +8
#pragma unroll
            for (int nt = 0; nt < 2; ++nt) {
                acc[mt][nt] = __builtin_amdgcn_wmma_f32_16x16x32_bf16(
                    false, af.v, false, bfrag[nt].v,
                    (short)0, acc[mt][nt], false, false);
            }
        }
        __syncthreads();              // protect buffer reuse next iteration
        cur ^= 1;
    }

    // ---- epilogue: C/D layout VGPR r: lanes0-15 M=r, lanes16-31 M=r+8 ----
#pragma unroll
    for (int mt = 0; mt < 4; ++mt)
#pragma unroll
        for (int nt = 0; nt < 2; ++nt)
#pragma unroll
            for (int r = 0; r < 8; ++r) {
                int row = m0 + waveM * 64 + mt * 16 + ks * 8 + r;
                int col = n0 + waveN * 32 + nt * 16 + am;
                if (col < N) {
                    float v = acc[mt][nt][r] + bias[col];
                    size_t o = (size_t)row * ldo + col;
                    if (mode == 0) {
                        v = fmaxf(v, 0.0f);
                        outB[o] = (__bf16)v;
                    } else if (mode == 1) {
                        v = fminf(fmaxf(v + 1.0f, 0.05f), 1e9f);
                        outF[o] = v;
                        outB[o] = (__bf16)v;
                    } else {
                        outF[o] = v;
                    }
                }
            }
}

// ---------------------------------------------------------------------------
// Softmax over K=2 + weighted combine: alpha|beta -> d_out
// ---------------------------------------------------------------------------
__global__ __launch_bounds__(256) void combine_kernel(
    const float* __restrict__ logits,   // [16384, 400]
    const float* __restrict__ Ef,       // [16384, 800]  (alpha|beta per row)
    float* __restrict__ out)            // alpha [8192,400] then beta [8192,400]
{
    int i = blockIdx.x * 256 + threadIdx.x;
    if (i >= 8192 * 400) return;
    int b = i / 400, j = i - b * 400;
    float l1 = logits[(size_t)b * 400 + j];
    float l2 = logits[(size_t)(b + 8192) * 400 + j];
    float m  = fmaxf(l1, l2);
    float w1 = __expf(l1 - m), w2 = __expf(l2 - m);
    float inv = 1.0f / (w1 + w2);
    w1 *= inv; w2 *= inv;
    float a1 = Ef[(size_t)b * 800 + j];
    float a2 = Ef[(size_t)(b + 8192) * 800 + j];
    float b1 = Ef[(size_t)b * 800 + 400 + j];
    float b2 = Ef[(size_t)(b + 8192) * 800 + 400 + j];
    out[(size_t)b * 400 + j]                 = w1 * a1 + w2 * a2;
    out[(size_t)(8192 * 400) + b * 400 + j]  = w1 * b1 + w2 * b2;
}

// ---------------------------------------------------------------------------
extern "C" void kernel_launch(void* const* d_in, const int* in_sizes, int n_in,
                              void* d_out, int out_size, void* d_ws, size_t ws_size,
                              hipStream_t stream) {
    (void)in_sizes; (void)n_in; (void)out_size; (void)ws_size;
    const int* a1  = (const int*)d_in[0];
    const int* r1  = (const int*)d_in[1];
    const int* a2  = (const int*)d_in[2];
    const int* r2  = (const int*)d_in[3];
    const float* ent = (const float*)d_in[4];
    const float* rel = (const float*)d_in[5];
    const float* pW1 = (const float*)d_in[6];
    const float* pb1 = (const float*)d_in[7];
    const float* pW2 = (const float*)d_in[8];
    const float* pb2 = (const float*)d_in[9];
    const float* pW0 = (const float*)d_in[10];
    const float* pb0 = (const float*)d_in[11];
    const float* iW1 = (const float*)d_in[12];
    const float* ib1 = (const float*)d_in[13];
    const float* iW2 = (const float*)d_in[14];
    const float* ib2 = (const float*)d_in[15];
    float* out = (float*)d_out;

    const int M = 16384;                       // 2 branches x 8192
    const size_t ACT = (size_t)M * 1600 * sizeof(__bf16);   // 52,428,800 B

    char* ws = (char*)d_ws;
    __bf16* Xc = (__bf16*)(ws);                // [M,1216] bf16
    __bf16* Y1 = (__bf16*)(ws + ACT);          // [M,1600] bf16
    __bf16* Y2 = (__bf16*)(ws + 2 * ACT);      // [M,1600] bf16
    float*  Ef = (float*)(ws);                 // [M,800] f32   (reuses Xc)
    __bf16* Eb = (__bf16*)(ws + ACT);          // [M,800] bf16  (reuses Y1)
    __bf16* Hb = (__bf16*)(ws + 2 * ACT);      // [M,800] bf16  (reuses Y2)
    float*  Lg = (float*)(ws + ACT);           // [M,400] f32   (reuses Eb)

    char* wts = ws + 3 * ACT;
    __bf16* W1b  = (__bf16*)(wts);                               // 1600x1216
    __bf16* W2b  = (__bf16*)(wts + 3891200);                     // 1600x1600
    __bf16* W0b  = (__bf16*)(wts + 3891200 + 5120000);           //  800x1600
    __bf16* iW1b = (__bf16*)(wts + 3891200 + 5120000 + 2560000); //  800x800
    __bf16* iW2b = (__bf16*)(wts + 3891200 + 5120000 + 2560000 + 1280000); // 400x800

    // 1) gather + regularize + concat (bf16, K-padded)
    prep_kernel<<<dim3(M), 256, 0, stream>>>(a1, r1, a2, r2, ent, rel, Xc);

    // 2) weight conversion (bf16, K-padded where needed)
    wconv_kernel<<<dim3(1600), 256, 0, stream>>>(pW1, W1b, 1200, 1216);
    wconv_kernel<<<dim3(1600), 256, 0, stream>>>(pW2, W2b, 1600, 1600);
    wconv_kernel<<<dim3(800),  256, 0, stream>>>(pW0, W0b, 1600, 1600);
    wconv_kernel<<<dim3(800),  256, 0, stream>>>(iW1, iW1b, 800, 800);
    wconv_kernel<<<dim3(400),  256, 0, stream>>>(iW2, iW2b, 800, 800);

    const dim3 blk(256);
    // 3) projection layer1: Y1 = relu(Xc @ pW1^T + pb1)   [M,1600]
    gemm_bf16_wmma<<<dim3(13, M / BM), blk, 0, stream>>>(
        Xc, 1216, W1b, 1216, pb1, 1600, 1216, 0, Y1, nullptr, 1600);
    // 4) projection layer2: Y2 = relu(Y1 @ pW2^T + pb2)   [M,1600]
    gemm_bf16_wmma<<<dim3(13, M / BM), blk, 0, stream>>>(
        Y1, 1600, W2b, 1600, pb2, 1600, 1600, 0, Y2, nullptr, 1600);
    // 5) projection layer0 + regularizer: E = reg(Y2 @ pW0^T + pb0) [M,800]
    gemm_bf16_wmma<<<dim3(7, M / BM), blk, 0, stream>>>(
        Y2, 1600, W0b, 1600, pb0, 800, 1600, 1, Eb, Ef, 800);
    // 6) intersection layer1: H = relu(E @ iW1^T + ib1)   [M,800]
    gemm_bf16_wmma<<<dim3(7, M / BM), blk, 0, stream>>>(
        Eb, 800, iW1b, 800, ib1, 800, 800, 0, Hb, nullptr, 800);
    // 7) intersection layer2: logits = H @ iW2^T + ib2    [M,400]
    gemm_bf16_wmma<<<dim3(4, M / BM), blk, 0, stream>>>(
        Hb, 800, iW2b, 800, ib2, 400, 800, 2, nullptr, Lg, 400);

    // 8) softmax over K=2 + weighted combine -> alpha|beta
    combine_kernel<<<dim3((8192 * 400 + 255) / 256), 256, 0, stream>>>(Lg, Ef, out);
}
